// ProjectedAttention_2224793059903
// MI455X (gfx1250) — matrix operand order repaired and checked by reading, not compiled, not run on hardware
//
#include <hip/hip_runtime.h>

typedef __attribute__((ext_vector_type(16))) _Float16 v16h;
typedef __attribute__((ext_vector_type(8)))  _Float16 v8h;
typedef __attribute__((ext_vector_type(8)))  float    v8f;

#define BATCH 32
#define SEQ   4096
#define DIM   128
#define TCHUNK 96               /* t-rows per block; 8 waves = 4 n-pairs x 2 t-halves */
#define FXSTAT 1048576.0f       /* 2^20 fixed-point scale for BN stats  */
#define FXSOFT 4294967296.0f    /* 2^32 fixed-point scale for softmax   */

// ---------- deterministic float reductions via integer atomics ----------
__device__ __forceinline__ unsigned fkey(float f) {
  unsigned b = __float_as_uint(f);
  return (b & 0x80000000u) ? ~b : (b | 0x80000000u);   // monotone float->uint
}
__device__ __forceinline__ float funkey(unsigned u) {
  unsigned b = (u & 0x80000000u) ? (u ^ 0x80000000u) : ~u;
  return __uint_as_float(b);
}
__device__ __forceinline__ void atomAddFx(long long* p, float v, float scale) {
  long long q = llrintf(v * scale);
  atomicAdd(reinterpret_cast<unsigned long long*>(p),
            static_cast<unsigned long long>(q));       // two's-complement wrap = signed add
}

// ---------- workspace init (harness poisons ws with 0xAA every run) ----------
__global__ void init_ws_kernel(long long* sums, long long* S0, long long* S1,
                               unsigned* maxK, unsigned* minK) {
  int i = blockIdx.x * 256 + threadIdx.x;
  if (i < 3 * 4 * DIM) sums[i] = 0;
  if (i < 3 * BATCH * DIM) { S0[i] = 0; S1[i] = 0; maxK[i] = 0u; minK[i] = 0xFFFFFFFFu; }
}

// ---------- one-time weight repack: Wh[c][tau*128+i] = (f16) W[c][i][tau] ----------
// Makes K contiguous per channel; each lane's B fragment is two 16B loads, 32B apart.
__global__ void repack_w_kernel(const float* __restrict__ Wv, const float* __restrict__ Wa,
                                _Float16* __restrict__ outV, _Float16* __restrict__ outA, int k) {
  const int n = DIM * DIM * k;
  for (int idx = blockIdx.x * 256 + threadIdx.x; idx < n; idx += gridDim.x * 256) {
    int c   = idx / (DIM * k);
    int rem = idx - c * (DIM * k);
    int tau = rem >> 7;
    int i   = rem & (DIM - 1);
    outV[idx] = (_Float16)Wv[(c * DIM + i) * k + tau];
    outA[idx] = (_Float16)Wa[(c * DIM + i) * k + tau];
  }
}

// ---------- fused dual-conv WMMA kernel (PASS 1 = stats, PASS 2 = softmax sums) ----------
template<int KSZ, int PASS>
__global__ void __launch_bounds__(256)
conv_wmma_kernel(const float* __restrict__ xval, const float* __restrict__ xatt,
                 const _Float16* __restrict__ Whv, const _Float16* __restrict__ Wha,
                 int P, int T,
                 long long* __restrict__ sums,      // [4][128]  (PASS 1)
                 unsigned*  __restrict__ maxK,      // [32][128] (PASS 1)
                 unsigned*  __restrict__ minK,      // [32][128] (PASS 1)
                 const float* __restrict__ alphaA,  // [128]     (PASS 2)
                 const float* __restrict__ rstar,   // [32][128] (PASS 2)
                 long long* __restrict__ S0,        // [32][128] (PASS 2)
                 long long* __restrict__ S1)        // [32][128] (PASS 2)
{
  constexpr int NR = TCHUNK + KSZ - 1;           // rows of padded x needed (<= 99)
  __shared__ _Float16 ldsV[NR][136];             // +16B row pad -> conflict-free b128 reads
  __shared__ _Float16 ldsA[NR][136];

  const int b  = blockIdx.y;
  const int t0 = blockIdx.x * TCHUNK;

  // stage x (fp32 -> fp16) for both streams; zero out-of-range (padding) rows
  for (int i = threadIdx.x; i < NR * DIM; i += 256) {
    int r = i >> 7, ch = i & (DIM - 1);
    int src = t0 + r - P;                        // padded row t0+r maps to x row t0+r-P
    float fv = 0.f, fa = 0.f;
    if (src >= 0 && src < SEQ) {
      size_t off = ((size_t)b * SEQ + src) * DIM + ch;
      fv = xval[off];
      fa = xatt[off];
    }
    ldsV[r][ch] = (_Float16)fv;
    ldsA[r][ch] = (_Float16)fa;
  }
  __syncthreads();

  const int lane  = threadIdx.x & 31;
  const int wave  = threadIdx.x >> 5;
  const int ln    = lane & 15;                   // M (A) / N (B,C,D) index
  const int hi    = (lane >> 4) << 3;            // 0 or 8: A K-half + C/D row-half offset
  const int kbase = (lane < 16) ? 0 : 8;         // B K-interleave matching A: low lanes K0..7,16..23, high K8..15,24..31
  const int wp    = wave & 3;                    // n-pair: channels wp*32 .. wp*32+31
  const int mb    = (wave >> 2) * 48;            // t-half base within the 96-row chunk

  v8f accV[2][3] = {}; v8f accA[2][3] = {};

  // K = tau*128 + i, stepped in 32-wide blocks: 4*KSZ WMMA k-steps
  for (int s = 0; s < 4 * KSZ; ++s) {
    const int tau  = s >> 2;
    const int cblk = (s & 3) * 32;
    // B fragments for both owned n-tiles: two 16-byte loads per lane, 32 bytes apart (packed fp16)
    v16h bV[2], bA[2];
#pragma unroll
    for (int u = 0; u < 2; ++u) {
      const size_t woff = (size_t)(wp * 32 + u * 16 + ln) * (KSZ * DIM) + (s << 5) + kbase;
      v8h b0 = *(const v8h*)(Whv + woff), b1 = *(const v8h*)(Whv + woff + 16);
      v8h c0 = *(const v8h*)(Wha + woff), c1 = *(const v8h*)(Wha + woff + 16);
      bV[u] = __builtin_shufflevector(b0, b1, 0,1,2,3,4,5,6,7,8,9,10,11,12,13,14,15);
      bA[u] = __builtin_shufflevector(c0, c1, 0,1,2,3,4,5,6,7,8,9,10,11,12,13,14,15);
    }
#pragma unroll
    for (int m = 0; m < 3; ++m) {                // three 16-row t sub-tiles per t-half
      const int row = mb + m * 16 + ln + tau;    // A[M=ln, K] = xpad[t0+mb+M+tau, i]
      const _Float16* pv = &ldsV[row][cblk + hi];
      const _Float16* pa = &ldsA[row][cblk + hi];
      v8h v0 = *(const v8h*)pv, v1 = *(const v8h*)(pv + 16);   // ISA A K-interleave
      v8h a0 = *(const v8h*)pa, a1 = *(const v8h*)(pa + 16);
      v16h aV = __builtin_shufflevector(v0, v1, 0,1,2,3,4,5,6,7,8,9,10,11,12,13,14,15);
      v16h aA = __builtin_shufflevector(a0, a1, 0,1,2,3,4,5,6,7,8,9,10,11,12,13,14,15);
#pragma unroll
      for (int u = 0; u < 2; ++u) {              // 1 ds_load_b128 per WMMA
        accV[u][m] = __builtin_amdgcn_wmma_f32_16x16x32_f16(false, aV, false, bV[u], (short)0, accV[u][m], false, false);
        accA[u][m] = __builtin_amdgcn_wmma_f32_16x16x32_f16(false, aA, false, bA[u], (short)0, accA[u][m], false, false);
      }
    }
  }

  // C/D layout: lane holds channel c; vgpr r of tile m is t = t0 + mb + m*16 + hi + r
  if constexpr (PASS == 1) {
#pragma unroll
    for (int u = 0; u < 2; ++u) {
      const int c = wp * 32 + u * 16 + ln;
      float sv = 0.f, qv = 0.f, sa = 0.f, qa = 0.f;
      float mx = -3.402823e38f, mn = 3.402823e38f;
#pragma unroll
      for (int m = 0; m < 3; ++m)
#pragma unroll
        for (int r = 0; r < 8; ++r) {
          int t = t0 + mb + m * 16 + hi + r;
          if (t < T) {
            float yv = accV[u][m][r], ya = accA[u][m][r];
            sv += yv; qv += yv * yv;
            sa += ya; qa += ya * ya;
            mx = fmaxf(mx, ya); mn = fminf(mn, ya);
          }
        }
      sv += __shfl_xor(sv, 16, 32); qv += __shfl_xor(qv, 16, 32);
      sa += __shfl_xor(sa, 16, 32); qa += __shfl_xor(qa, 16, 32);
      mx = fmaxf(mx, __shfl_xor(mx, 16, 32));
      mn = fminf(mn, __shfl_xor(mn, 16, 32));
      if (lane < 16) {
        atomAddFx(&sums[c],            sv, FXSTAT);
        atomAddFx(&sums[DIM + c],      qv, FXSTAT);
        atomAddFx(&sums[2 * DIM + c],  sa, FXSTAT);
        atomAddFx(&sums[3 * DIM + c],  qa, FXSTAT);
        atomicMax(&maxK[b * DIM + c], fkey(mx));
        atomicMin(&minK[b * DIM + c], fkey(mn));
      }
    }
  } else {
#pragma unroll
    for (int u = 0; u < 2; ++u) {
      const int c = wp * 32 + u * 16 + ln;
      float aA_ = alphaA[c];
      float rs  = rstar[b * DIM + c];
      float s0 = 0.f, s1 = 0.f;
#pragma unroll
      for (int m = 0; m < 3; ++m)
#pragma unroll
        for (int r = 0; r < 8; ++r) {
          int t = t0 + mb + m * 16 + hi + r;
          if (t < T) {
            float w = __expf(aA_ * accA[u][m][r] - rs);   // max-shifted softmax weight
            s0 += w;
            s1 += w * accV[u][m][r];
          }
        }
      s0 += __shfl_xor(s0, 16, 32);
      s1 += __shfl_xor(s1, 16, 32);
      if (lane < 16) {
        atomAddFx(&S0[b * DIM + c], s0, FXSOFT);
        atomAddFx(&S1[b * DIM + c], s1, FXSOFT);
      }
    }
  }
}

// ---------- per-branch BN/softmax parameter finalization ----------
__global__ void finalize_kernel(const long long* __restrict__ sums,
                                const unsigned* __restrict__ maxK, const unsigned* __restrict__ minK,
                                const float* __restrict__ gv, const float* __restrict__ betav,
                                const float* __restrict__ ga,
                                float* __restrict__ alphaA, float* __restrict__ alphaV,
                                float* __restrict__ cV, float* __restrict__ rstar, int T) {
  int i = blockIdx.x * 256 + threadIdx.x;
  if (i >= BATCH * DIM) return;
  int b = i >> 7, c = i & (DIM - 1);
  double n  = 32.0 * (double)T;
  double sV = (double)sums[c]           / (double)FXSTAT;
  double qV = (double)sums[DIM + c]     / (double)FXSTAT;
  double sA = (double)sums[2 * DIM + c] / (double)FXSTAT;
  double qA = (double)sums[3 * DIM + c] / (double)FXSTAT;
  float mV = (float)(sV / n);
  float vV = (float)(qV / n) - mV * mV;
  float aV = gv[c] * rsqrtf(vV + 1e-5f);
  float mA = (float)(sA / n);
  float vA = (float)(qA / n) - mA * mA;
  float aA = ga[c] * rsqrtf(vA + 1e-5f);
  if (b == 0) {                       // conv bias cancels inside BN; beta_a cancels in softmax
    alphaA[c] = aA;
    alphaV[c] = aV;
    cV[c]     = betav[c] - aV * mV;
  }
  float ymax = funkey(maxK[i]);
  float ymin = funkey(minK[i]);
  rstar[i] = (aA >= 0.0f) ? aA * ymax : aA * ymin;   // max_t of alpha_a * y_a
}

// ---------- output: out[b, j*D+c] = alpha_v * S1/S0 + (beta_v - alpha_v*mean_v) ----------
__global__ void output_kernel(const long long* __restrict__ S0, const long long* __restrict__ S1,
                              const float* __restrict__ alphaV, const float* __restrict__ cV,
                              float* __restrict__ out) {
  int i = blockIdx.x * 256 + threadIdx.x;
  if (i >= BATCH * 3 * DIM) return;
  int b  = i / (3 * DIM);
  int rj = i - b * 3 * DIM;
  int j  = rj >> 7;
  int c  = rj & (DIM - 1);
  int idx = j * BATCH * DIM + b * DIM + c;
  double s0 = (double)S0[idx] / (double)FXSOFT;
  double s1 = (double)S1[idx] / (double)FXSOFT;
  out[i] = alphaV[j * DIM + c] * (float)(s1 / s0) + cV[j * DIM + c];
}

extern "C" void kernel_launch(void* const* d_in, const int* in_sizes, int n_in,
                              void* d_out, int out_size, void* d_ws, size_t ws_size,
                              hipStream_t stream) {
  (void)in_sizes; (void)n_in; (void)out_size; (void)ws_size;
  const float* xv = (const float*)d_in[0];
  const float* xa = (const float*)d_in[1];
  const float* Wv[3]; const float* Wa[3];
  const float* gv[3]; const float* bev[3]; const float* ga[3];
  for (int j = 0; j < 3; ++j) {
    int base = 2 + j * 8;                       // Wv, bv, Wa, ba, gv, betav, ga, betaa
    Wv[j]  = (const float*)d_in[base + 0];
    Wa[j]  = (const float*)d_in[base + 2];
    gv[j]  = (const float*)d_in[base + 4];
    bev[j] = (const float*)d_in[base + 5];
    ga[j]  = (const float*)d_in[base + 6];
  }

  const int kj[3] = {2, 3, 4};
  const int Tj[3] = {4096, 4097, 4099};   // conv output length per branch
  const int Pj[3] = {1, 3, 6};            // cumulative left padding per branch
  const int Cj[3] = {43, 43, 43};         // ceil(T/96)

  // ---- workspace layout (~950 KB total) ----
  char* p = (char*)d_ws;
  long long* sums = (long long*)p;  p += (size_t)3 * 4 * DIM * 8;          // [br][4][128]
  unsigned*  maxK = (unsigned*)p;   p += (size_t)3 * BATCH * DIM * 4;      // [br][32][128]
  unsigned*  minK = (unsigned*)p;   p += (size_t)3 * BATCH * DIM * 4;
  long long* S0   = (long long*)p;  p += (size_t)3 * BATCH * DIM * 8;
  long long* S1   = (long long*)p;  p += (size_t)3 * BATCH * DIM * 8;
  float* alphaA   = (float*)p;      p += (size_t)3 * DIM * 4;
  float* alphaV   = (float*)p;      p += (size_t)3 * DIM * 4;
  float* cV       = (float*)p;      p += (size_t)3 * DIM * 4;
  float* rstar    = (float*)p;      p += (size_t)3 * BATCH * DIM * 4;
  _Float16* Whv[3]; _Float16* Wha[3];
  for (int j = 0; j < 3; ++j) {                 // packed fp16 weights, fragment-ordered
    Whv[j] = (_Float16*)p; p += (size_t)DIM * DIM * kj[j] * 2;
    Wha[j] = (_Float16*)p; p += (size_t)DIM * DIM * kj[j] * 2;
  }

  init_ws_kernel<<<48, 256, 0, stream>>>(sums, S0, S1, maxK, minK);
  for (int j = 0; j < 3; ++j)
    repack_w_kernel<<<64, 256, 0, stream>>>(Wv[j], Wa[j], Whv[j], Wha[j], kj[j]);

  // ---- pass 1: conv (WMMA) + deterministic BN stats & row extrema ----
  conv_wmma_kernel<2,1><<<dim3(Cj[0], BATCH), 256, 0, stream>>>(xv, xa, Whv[0], Wha[0], Pj[0], Tj[0],
      sums + 0 * 512, maxK + 0 * 4096, minK + 0 * 4096, nullptr, nullptr, nullptr, nullptr);
  conv_wmma_kernel<3,1><<<dim3(Cj[1], BATCH), 256, 0, stream>>>(xv, xa, Whv[1], Wha[1], Pj[1], Tj[1],
      sums + 1 * 512, maxK + 1 * 4096, minK + 1 * 4096, nullptr, nullptr, nullptr, nullptr);
  conv_wmma_kernel<4,1><<<dim3(Cj[2], BATCH), 256, 0, stream>>>(xv, xa, Whv[2], Wha[2], Pj[2], Tj[2],
      sums + 2 * 512, maxK + 2 * 4096, minK + 2 * 4096, nullptr, nullptr, nullptr, nullptr);

  // ---- finalize BN affine params + per-(b,c) softmax shift ----
  for (int j = 0; j < 3; ++j)
    finalize_kernel<<<16, 256, 0, stream>>>(sums + j * 512, maxK + j * 4096, minK + j * 4096,
        gv[j], bev[j], ga[j], alphaA + j * DIM, alphaV + j * DIM, cV + j * DIM,
        rstar + j * 4096, Tj[j]);

  // ---- pass 2: recompute conv (WMMA), accumulate softmax numerator/denominator ----
  conv_wmma_kernel<2,2><<<dim3(Cj[0], BATCH), 256, 0, stream>>>(xv, xa, Whv[0], Wha[0], Pj[0], Tj[0],
      nullptr, nullptr, nullptr, alphaA + 0 * DIM, rstar + 0 * 4096, S0 + 0 * 4096, S1 + 0 * 4096);
  conv_wmma_kernel<3,2><<<dim3(Cj[1], BATCH), 256, 0, stream>>>(xv, xa, Whv[1], Wha[1], Pj[1], Tj[1],
      nullptr, nullptr, nullptr, alphaA + 1 * DIM, rstar + 1 * 4096, S0 + 1 * 4096, S1 + 1 * 4096);
  conv_wmma_kernel<4,2><<<dim3(Cj[2], BATCH), 256, 0, stream>>>(xv, xa, Whv[2], Wha[2], Pj[2], Tj[2],
      nullptr, nullptr, nullptr, alphaA + 2 * DIM, rstar + 2 * 4096, S0 + 2 * 4096, S1 + 2 * 4096);

  output_kernel<<<48, 256, 0, stream>>>(S0, S1, alphaV, cV, (float*)d_out);
}
